// VanillaRNNBlock_18356690223549
// MI455X (gfx1250) — matrix operand order repaired and checked by reading, not compiled, not run on hardware
//
#include <hip/hip_runtime.h>
#include <hip/hip_bf16.h>
#include <math.h>

// ---------------------------------------------------------------------------
// VanillaRNNBlock on MI455X (gfx1250, wave32, WMMA, TDM)
//   d_out = outputs (B,S,H) fp32  ++  h_last (B,H) fp32
// Streaming (>L2-sized) traffic uses non-temporal hints so the 192 MB L2
// keeps the reused working set (weights, h state) resident.
// ---------------------------------------------------------------------------

typedef __attribute__((ext_vector_type(16))) __bf16 v16bf;
typedef __attribute__((ext_vector_type(8)))  float  v8f;
typedef __attribute__((ext_vector_type(4)))  float  f32x4;   // clang vec (NT-builtin ok)
typedef __attribute__((ext_vector_type(4)))  unsigned int u32x4;
typedef __attribute__((ext_vector_type(8)))  int i32x8;
typedef __attribute__((ext_vector_type(4)))  int i32x4;

union BF16Tile { v16bf v; uint4 u[2]; };   // 32 bytes = 8 VGPRs per lane

#define RNN_B 64
#define RNN_S 2048
#define RNN_I 512
#define RNN_H 512
#define SCAN_NWG 8

// LDS row layout: 512 bf16 (1024 B) + 16 B TDM pad -> 1040 B row stride.
// Stride = 260 words -> consecutive rows advance 4 banks: conflict-minimal
// for 16-lane x b128 reads.
#define LDS_ROW_BYTES 1040
#define XPROJ_COLS 128                       // B-panel columns staged per block
#define XPROJ_LDS  (XPROJ_COLS * LDS_ROW_BYTES)   // 133,120 B
#define SCAN_LDS   (RNN_B * LDS_ROW_BYTES)        //  66,560 B

__device__ __forceinline__ v8f wmma_bf16(v16bf a, v16bf b, v8f c) {
  return __builtin_amdgcn_wmma_f32_16x16x32_bf16(
      false, a, false, b, (short)0, c, false, false);
}

// ---------------------------------------------------------------------------
// TDM: async-copy a 2D bf16 tile (tile_d0 elems/row x tile_d1 rows, row
// stride stride0 elems) from global to LDS, inserting 16 B of padding after
// every 1024 B row (pad_interval=7 => 256 DWORDs, pad_amount=3 => 4 DWORDs).
// D# fields per CDNA5 ISA. Issue from ONE wave.
// ---------------------------------------------------------------------------
__device__ __forceinline__ void tdm_load_2d_bf16(unsigned lds_off, const void* gsrc,
                                                 unsigned tile_d0, unsigned tile_d1,
                                                 unsigned tensor_d0, unsigned tensor_d1,
                                                 unsigned stride0) {
  unsigned long long ga = (unsigned long long)(size_t)gsrc;
  u32x4 g0;
  g0[0] = 1u;                                              // count=1, user desc
  g0[1] = lds_off;                                         // LDS byte address
  g0[2] = (unsigned)ga;                                    // global_addr[31:0]
  g0[3] = (unsigned)((ga >> 32) & 0x01FFFFFFu) | (2u << 30); // addr[56:32] | type=2
  i32x8 g1;
  g1[0] = (int)((1u << 16) |        // data_size = 1 (2 bytes)
                (1u << 20) |        // pad_enable
                (7u << 22) |        // pad_interval: 256 DWORDs
                (3u << 25));        // pad_amount:   4 DWORDs (16 B)
  g1[1] = (int)((tensor_d0 & 0xFFFFu) << 16);                       // dim0[15:0]
  g1[2] = (int)(((tensor_d0 >> 16) & 0xFFFFu) | ((tensor_d1 & 0xFFFFu) << 16));
  g1[3] = (int)(((tensor_d1 >> 16) & 0xFFFFu) | ((tile_d0 & 0xFFFFu) << 16));
  g1[4] = (int)(tile_d1 & 0xFFFFu);                                 // tile_dim1
  g1[5] = (int)stride0;                                             // dim0_stride lo
  g1[6] = 0;
  g1[7] = 0;
  i32x4 z4 = {0, 0, 0, 0};
#if __has_include(<hip/amd_detail/amd_gfx1250_TDM.h>)
  i32x8 z8 = {0, 0, 0, 0, 0, 0, 0, 0};
  __builtin_amdgcn_tensor_load_to_lds(g0, g1, z4, z4, z8, 0);
#else
  __builtin_amdgcn_tensor_load_to_lds(g0, g1, z4, z4, 0);
#endif
}

// ---------------------------------------------------------------------------
// Kernel 1: fp32 -> bf16 weight/state conversion + barrier reset each launch.
// ---------------------------------------------------------------------------
__global__ void rnn_prep(const float* __restrict__ Wx, const float* __restrict__ Wh,
                         const float* __restrict__ h0,
                         __bf16* __restrict__ Wxb, __bf16* __restrict__ Whb,
                         __bf16* __restrict__ h0b, unsigned* __restrict__ bar) {
  const int tid    = blockIdx.x * blockDim.x + threadIdx.x;
  const int stride = gridDim.x * blockDim.x;
  for (int i = tid; i < RNN_H * RNN_I; i += stride) {
    Wxb[i] = (__bf16)Wx[i];
    Whb[i] = (__bf16)Wh[i];
  }
  for (int i = tid; i < RNN_B * RNN_H; i += stride) h0b[i] = (__bf16)h0[i];
  if (tid == 0) { bar[0] = 0u; bar[1] = 0u; }
}

// ---------------------------------------------------------------------------
// Kernel 2: xproj = x @ Wx^T -> fp32, written in-place into d_out outputs.
// Block = 8 waves; TDM stages a 128-col B panel (Wx rows) into padded LDS
// once; each wave computes a 16(M) x 128(N) strip; A (x fp32, NT-stream) is
// converted on the fly; B tiles come from LDS via ds_load_b128.
// ---------------------------------------------------------------------------
__global__ void __launch_bounds__(256) xproj_gemm(const float* __restrict__ x,
                                                  const __bf16* __restrict__ Wxb,
                                                  float* __restrict__ out) {
  extern __shared__ char smem[];
  const int lane    = threadIdx.x & 31;
  const int wave    = threadIdx.x >> 5;
  const int colgrp  = blockIdx.x & 3;           // 4 col groups of 128
  const int mgrp    = blockIdx.x >> 2;          // 1024 groups of 8 m-tiles
  const int mtile   = mgrp * 8 + wave;          // 0..8191
  const int rowbase = mtile * 16;               // flattened (b*S+s) row
  const int mrow    = lane & 15;
  const int o       = (lane < 16) ? 0 : 8;      // A-operand K sub-offset
  const int koffsel = (lane < 16) ? 0 : 16;     // B-operand K half
  const int colbase = colgrp * XPROJ_COLS;

  // ---- TDM: stage Wx[colbase .. colbase+127][0..511] into padded LDS
  if (threadIdx.x < 32) {
    tdm_load_2d_bf16((unsigned)(size_t)smem, Wxb + (size_t)colbase * RNN_I,
                     RNN_I, XPROJ_COLS, RNN_I, RNN_H, RNN_I);
    __builtin_amdgcn_s_wait_tensorcnt(0);
  }
  __syncthreads();

  const v8f zero8 = {0.f, 0.f, 0.f, 0.f, 0.f, 0.f, 0.f, 0.f};
  v8f acc[8];
#pragma unroll
  for (int nt = 0; nt < 8; ++nt) acc[nt] = zero8;

  const float* xrow = x + (size_t)(rowbase + mrow) * RNN_I;

  for (int kt = 0; kt < 16; ++kt) {
    const int kb = kt * 32;
    // ---- A tile: 16 fp32 -> bf16, non-temporal (x is a 256 MB read-once stream)
    f32x4 a0 = __builtin_nontemporal_load((const f32x4*)(xrow + kb + o));
    f32x4 a1 = __builtin_nontemporal_load((const f32x4*)(xrow + kb + o + 4));
    f32x4 a2 = __builtin_nontemporal_load((const f32x4*)(xrow + kb + 16 + o));
    f32x4 a3 = __builtin_nontemporal_load((const f32x4*)(xrow + kb + 16 + o + 4));
    v16bf a;
    a[0]  = (__bf16)a0[0]; a[1]  = (__bf16)a0[1]; a[2]  = (__bf16)a0[2]; a[3]  = (__bf16)a0[3];
    a[4]  = (__bf16)a1[0]; a[5]  = (__bf16)a1[1]; a[6]  = (__bf16)a1[2]; a[7]  = (__bf16)a1[3];
    a[8]  = (__bf16)a2[0]; a[9]  = (__bf16)a2[1]; a[10] = (__bf16)a2[2]; a[11] = (__bf16)a2[3];
    a[12] = (__bf16)a3[0]; a[13] = (__bf16)a3[1]; a[14] = (__bf16)a3[2]; a[15] = (__bf16)a3[3];

    // ---- 8 n-tiles reuse this A tile; B from LDS (low latency, pipelinable)
#pragma unroll
    for (int nt = 0; nt < 8; ++nt) {
      const int c = nt * 16 + (lane & 15);            // local column 0..127
      const char* bp = smem + (size_t)c * LDS_ROW_BYTES + (size_t)(kb + koffsel / 2) * 2;
      BF16Tile b;
      b.u[0] = *(const uint4*)bp;
      b.u[1] = *(const uint4*)(bp + 32);
      acc[nt] = wmma_bf16(a, b.v, acc[nt]);
    }
  }

  // ---- epilogue: C layout (VGPR r -> row r / r+8) -> row-major fp32, NT store
  const int rtop = (lane < 16) ? 0 : 8;
#pragma unroll
  for (int nt = 0; nt < 8; ++nt) {
    const int col = colbase + nt * 16 + (lane & 15);
#pragma unroll
    for (int r = 0; r < 8; ++r) {
      const int row = rowbase + rtop + r;
      __builtin_nontemporal_store(acc[nt][r], out + (size_t)row * RNN_H + col);
    }
  }
}

// ---------------------------------------------------------------------------
// Kernel 3: persistent scan. 8 WGs x 256 threads; each wave pins a 16-col
// slice of Wh in 128 VGPRs for all 2048 steps. Each step: TDM-copy the 64 KB
// bf16 h state into padded LDS, 32 WMMAs/wave from LDS A-tiles, fused
// tanh(+xproj+bias) epilogue in-place over d_out (NT, touched exactly once),
// then a device-wide sense-reversal barrier.
// ---------------------------------------------------------------------------
__global__ void __launch_bounds__(256) rnn_scan(const __bf16* __restrict__ Whb,
                                                const float* __restrict__ bh,
                                                float* __restrict__ out,
                                                __bf16* __restrict__ hbuf0,
                                                __bf16* __restrict__ hbuf1,
                                                unsigned* __restrict__ bar) {
  extern __shared__ char smem[];
  const int lane    = threadIdx.x & 31;
  const int wave    = threadIdx.x >> 5;
  const int ntile   = wave >> 1;                 // 0..3 (16 cols each)
  const int mhalf   = wave & 1;                  // rows 0..31 or 32..63
  const int mrow    = lane & 15;
  const int o       = (lane < 16) ? 0 : 8;
  const int koffsel = (lane < 16) ? 0 : 16;
  const int rtop    = (lane < 16) ? 0 : 8;
  const int col     = blockIdx.x * 64 + ntile * 16 + (lane & 15);

  // ---- pin this wave's Wh slice (16 cols x K=512) in registers
  v16bf btile[16];
#pragma unroll
  for (int kt = 0; kt < 16; ++kt) {
    const __bf16* bp = Whb + (size_t)col * RNN_H + kt * 32 + koffsel / 2;
    BF16Tile b;
    b.u[0] = *(const uint4*)bp;
    b.u[1] = *(const uint4*)(bp + 16);
    btile[kt] = b.v;
  }
  const float bias = bh[col];
  const v8f zero8 = {0.f, 0.f, 0.f, 0.f, 0.f, 0.f, 0.f, 0.f};

  __bf16* hc = hbuf0;
  __bf16* hn = hbuf1;

  for (int t = 0; t < RNN_S; ++t) {
    // ---- TDM: stage full h state (64 x 512 bf16) into padded LDS
    if (threadIdx.x < 32) {
      tdm_load_2d_bf16((unsigned)(size_t)smem, hc,
                       RNN_H, RNN_B, RNN_H, RNN_B, RNN_H);
      __builtin_amdgcn_s_wait_tensorcnt(0);
    }
    __syncthreads();

    v8f acc0 = zero8, acc1 = zero8;
    const char* h0row = smem + (size_t)(mhalf * 32 + mrow) * LDS_ROW_BYTES;
    const char* h1row = h0row + 16 * LDS_ROW_BYTES;
#pragma unroll
    for (int kt = 0; kt < 16; ++kt) {
      const char* p0 = h0row + kt * 64 + o * 2;   // kt*32 elems, +o elems
      const char* p1 = h1row + kt * 64 + o * 2;
      BF16Tile a0, a1;
      a0.u[0] = *(const uint4*)p0;
      a0.u[1] = *(const uint4*)(p0 + 32);         // +16 elems
      a1.u[0] = *(const uint4*)p1;
      a1.u[1] = *(const uint4*)(p1 + 32);
      acc0 = wmma_bf16(a0.v, btile[kt], acc0);
      acc1 = wmma_bf16(a1.v, btile[kt], acc1);
    }

    // ---- fused epilogue: h = tanh(xproj_t + h@Wh^T + b); in-place over xproj
#pragma unroll
    for (int mi = 0; mi < 2; ++mi) {
      v8f acc = mi ? acc1 : acc0;
#pragma unroll
      for (int r = 0; r < 8; ++r) {
        const int row = mhalf * 32 + mi * 16 + rtop + r;       // batch index
        const size_t oidx = ((size_t)row * RNN_S + t) * RNN_H + col;
        const float xp = __builtin_nontemporal_load(out + oidx);
        const float v = tanhf(acc[r] + xp + bias);
        __builtin_nontemporal_store(v, out + oidx);
        hn[(size_t)row * RNN_H + col] = (__bf16)v;             // RT: hot next step
        if (t == RNN_S - 1)
          out[(size_t)RNN_B * RNN_S * RNN_H + (size_t)row * RNN_H + col] = v;
      }
    }

    // ---- device-wide sense-reversal barrier between steps
    __threadfence();
    __syncthreads();
    if (threadIdx.x == 0) {
      unsigned g = __hip_atomic_load(&bar[1], __ATOMIC_RELAXED, __HIP_MEMORY_SCOPE_AGENT);
      unsigned a = __hip_atomic_fetch_add(&bar[0], 1u, __ATOMIC_ACQ_REL, __HIP_MEMORY_SCOPE_AGENT);
      if (a == SCAN_NWG - 1) {
        __hip_atomic_store(&bar[0], 0u, __ATOMIC_RELAXED, __HIP_MEMORY_SCOPE_AGENT);
        __hip_atomic_store(&bar[1], g + 1u, __ATOMIC_RELEASE, __HIP_MEMORY_SCOPE_AGENT);
      } else {
        while (__hip_atomic_load(&bar[1], __ATOMIC_ACQUIRE, __HIP_MEMORY_SCOPE_AGENT) == g) {
          __builtin_amdgcn_s_sleep(2);
        }
      }
    }
    __syncthreads();

    __bf16* tmp = hc; hc = hn; hn = tmp;
  }
}

// ---------------------------------------------------------------------------
// Host launcher. Workspace layout (bytes):
//   [0,256) barrier | [+512K) Wx bf16 | [+512K) Wh bf16 | 2 x 64K h buffers
// ---------------------------------------------------------------------------
extern "C" void kernel_launch(void* const* d_in, const int* in_sizes, int n_in,
                              void* d_out, int out_size, void* d_ws, size_t ws_size,
                              hipStream_t stream) {
  const float* x  = (const float*)d_in[0];
  const float* h0 = (const float*)d_in[1];
  const float* Wx = (const float*)d_in[2];
  const float* Wh = (const float*)d_in[3];
  const float* bh = (const float*)d_in[4];
  float* out = (float*)d_out;

  char* ws = (char*)d_ws;
  unsigned* bar = (unsigned*)ws;
  __bf16* Wxb = (__bf16*)(ws + 256);
  __bf16* Whb = (__bf16*)(ws + 256 + 512 * 1024);
  __bf16* h0b = (__bf16*)(ws + 256 + 1024 * 1024);
  __bf16* h1b = h0b + RNN_B * RNN_H;

  rnn_prep<<<256, 256, 0, stream>>>(Wx, Wh, h0, Wxb, Whb, h0b, bar);
  xproj_gemm<<<(RNN_B * RNN_S / 16 / 8) * 4, 256, XPROJ_LDS, stream>>>(x, Wxb, out);
  rnn_scan<<<SCAN_NWG, 256, SCAN_LDS, stream>>>(Whb, bh, out, h0b, h1b, bar);
}
